// Pipeline_5755256177385
// MI455X (gfx1250) — matrix operand order repaired and checked by reading, not compiled, not run on hardware
//
#include <hip/hip_runtime.h>

// ---------------- CDNA5 WMMA / TDM types ----------------
typedef float        v2f   __attribute__((ext_vector_type(2)));
typedef float        v8f   __attribute__((ext_vector_type(8)));
typedef __bf16       v16bf __attribute__((ext_vector_type(16)));
typedef unsigned int v4u   __attribute__((ext_vector_type(4)));
typedef int          v4i   __attribute__((ext_vector_type(4)));
typedef int          v8i_t __attribute__((ext_vector_type(8)));

// ---------------- problem constants ----------------
#define BZ     16
#define LSEQ   512
#define DB     768        // DBERT
#define NERD   17         // NER_DIM
#define NEDD   300        // NED_DIM
#define NKBC   100000
#define REDIM  32
#define HTC    128
#define EMAXC  32
#define KNNC   10
#define NHEADC 8
#define DINC   785
#define HDIMC  784
#define DHH_   98         // head dim
#define FFC    2048
#define NLAY   6
#define D3C    1085
#define SEQT   33         // 1 + EMAX
#define ROWS   (BZ*SEQT)          // 528
#define NERROWS (BZ*(LSEQ+1))     // 8208
#define KPAD   320        // NEDD padded to mult of 32 for bf16 WMMA

// KNN dynamic-LDS layout (kernel has no static LDS -> dynamic segment starts at 0)
#define KNN_QS_OFF    0
#define KNN_KBT_OFF   (16*KPAD*4)                   // 20480
#define KNN_BUF_BYTES (128*KPAD*2)                  // 81920 per buffer
#define KNN_D2S_OFF   (KNN_KBT_OFF + 2*KNN_BUF_BYTES)
#define KNN_MD_OFF    (KNN_D2S_OFF + 16*128*4)
#define KNN_MI_OFF    (KNN_MD_OFF + 16*160*4)
#define KNN_SHMEM     (KNN_MI_OFF + 16*160*4)       // 212992 bytes (needs 320KB WGP LDS)

// =====================================================================
// Generic GEMM:  C(MxN) = act( A(MxK) @ B(KxN) + bias )   f32, WMMA 16x16x4
// One wave computes one 16x16 tile; block = 4 waves.
// Interior tiles take an unguarded fast path (no exec-mask churn).
// =====================================================================
__launch_bounds__(128)
__global__ void gemm_wmma_f32(const float* __restrict__ A, const float* __restrict__ B,
                              const float* __restrict__ bias, float* __restrict__ C,
                              int M, int N, int K, int act)
{
    const int lane  = threadIdx.x & 31;
    const int wave  = threadIdx.x >> 5;
    const int tilesN = (N + 15) >> 4;
    const int tilesM = (M + 15) >> 4;
    long tile = (long)blockIdx.x * 4 + wave;
    if (tile >= (long)tilesM * tilesN) return;        // wave-uniform
    const int tm = (int)(tile / tilesN) * 16;
    const int tn = (int)(tile % tilesN) * 16;
    const int g  = lane >> 4;
    const int li = lane & 15;
    const int mi = tm + li;
    const int ni = tn + li;
    v8f acc = {};
    const bool interior = (tm + 16 <= M) && (tn + 16 <= N);
    if (interior) {
        const float* __restrict__ Ap = A + (long)mi * K;
        const int kMain = K & ~3;
        for (int k0 = 0; k0 < kMain; k0 += 4) {
            const int ka = k0 + 2 * g;
            v2f a, b;
            a.x = Ap[ka];
            a.y = Ap[ka + 1];
            b.x = B[(long)ka * N + ni];
            b.y = B[(long)(ka + 1) * N + ni];
            acc = __builtin_amdgcn_wmma_f32_16x16x4_f32(false, a, false, b, (short)0, acc, false, false);
        }
        if (kMain < K) {                               // K remainder (guarded once)
            const int ka = kMain + 2 * g;
            v2f a, b;
            a.x = (ka     < K) ? Ap[ka]                   : 0.f;
            a.y = (ka + 1 < K) ? Ap[ka + 1]               : 0.f;
            b.x = (ka     < K) ? B[(long)ka * N + ni]     : 0.f;
            b.y = (ka + 1 < K) ? B[(long)(ka + 1) * N + ni] : 0.f;
            acc = __builtin_amdgcn_wmma_f32_16x16x4_f32(false, a, false, b, (short)0, acc, false, false);
        }
    } else {
        for (int k0 = 0; k0 < K; k0 += 4) {
            const int ka = k0 + 2 * g;
            v2f a, b;
            a.x = (mi < M && ka     < K) ? A[(long)mi * K + ka]       : 0.f;
            a.y = (mi < M && ka + 1 < K) ? A[(long)mi * K + ka + 1]   : 0.f;
            b.x = (ni < N && ka     < K) ? B[(long)ka * N + ni]       : 0.f;
            b.y = (ni < N && ka + 1 < K) ? B[(long)(ka + 1) * N + ni] : 0.f;
            acc = __builtin_amdgcn_wmma_f32_16x16x4_f32(false, a, false, b, (short)0, acc, false, false);
        }
    }
#pragma unroll
    for (int i = 0; i < 8; ++i) {
        int row = tm + i + 8 * g;
        int col = tn + li;
        if (row < M && col < N) {
            float v = acc[i];
            if (bias) v += bias[col];
            if (act == 1) v = fmaxf(v, 0.f);
            C[(long)row * N + col] = v;
        }
    }
}

// =====================================================================
// NER softmax + xa build:  xa = [x | softmax(ner)], also emit ner_out
// =====================================================================
__launch_bounds__(256)
__global__ void xabuild_kernel(const float* __restrict__ x, const float* __restrict__ ner,
                               float* __restrict__ xa, float* __restrict__ nerOut)
{
    const int r = blockIdx.x;              // 0..8207
    const int b = r / (LSEQ + 1);
    const int s = r % (LSEQ + 1);
    __shared__ float sm[NERD];
    for (int d = threadIdx.x; d < DB; d += 256)
        xa[(long)r * DINC + d] = x[(long)r * DB + d];
    if (threadIdx.x < NERD) sm[threadIdx.x] = ner[(long)r * NERD + threadIdx.x];
    __syncthreads();
    if (threadIdx.x == 0) {
        float mx = sm[0];
        for (int c = 1; c < NERD; ++c) mx = fmaxf(mx, sm[c]);
        float su = 0.f;
        for (int c = 0; c < NERD; ++c) { sm[c] = __expf(sm[c] - mx); su += sm[c]; }
        float inv = 1.f / su;
        for (int c = 0; c < NERD; ++c) sm[c] *= inv;
    }
    __syncthreads();
    if (threadIdx.x < NERD) {
        xa[(long)r * DINC + DB + threadIdx.x] = sm[threadIdx.x];
        if (s >= 1)
            nerOut[((long)b * LSEQ + (s - 1)) * NERD + threadIdx.x] = ner[(long)r * NERD + threadIdx.x];
    }
}

// =====================================================================
// Entity filter (per-batch serial selection mimicking key-sorted order)
// =====================================================================
__launch_bounds__(512)
__global__ void entity_filter_kernel(const float* __restrict__ xa, float* __restrict__ ents,
                                     float* __restrict__ outPos, float* __restrict__ outValid)
{
    const int b = blockIdx.x;
    __shared__ int amax[LSEQ];
    __shared__ int slotKind[EMAXC], slotI[EMAXC], slotE[EMAXC];
    __shared__ int nsel;
    {
        int i = threadIdx.x;
        const float* pr = xa + ((long)(b * (LSEQ + 1) + 1 + i)) * DINC + DB;
        int am = 0; float bv = pr[0];
        for (int c = 1; c < NERD; ++c) if (pr[c] > bv) { bv = pr[c]; am = c; }
        amax[i] = am;
    }
    __syncthreads();
    if (threadIdx.x == 0) {
        int cnt = 0;
        for (int t = 0; t < 4 && cnt < EMAXC; ++t) {
            for (int i = 0; i < LSEQ && cnt < EMAXC; ++i)        // singles (S tag)
                if (amax[i] == 4 * t + 3) { slotKind[cnt] = 0; slotI[cnt] = i; slotE[cnt] = 0; ++cnt; }
            for (int i = 0; i < LSEQ && cnt < EMAXC; ++i) {      // begins (B tag)
                if (amax[i] == 4 * t) {
                    int end = LSEQ;
                    for (int j = i + 1; j < LSEQ; ++j) if (amax[j] == 4 * t + 2) { end = j; break; }
                    slotKind[cnt] = 1; slotI[cnt] = i; slotE[cnt] = end; ++cnt;
                }
            }
        }
        nsel = cnt;
    }
    __syncthreads();
    for (int s = 0; s < EMAXC; ++s) {
        const int v    = (s < nsel);
        const int kind = v ? slotKind[s] : 0;
        const int i0   = v ? slotI[s]    : 0;
        const int e0   = v ? slotE[s]    : 0;
        for (int d = threadIdx.x; d < DINC; d += 512) {
            float val = 0.f;
            if (v) {
                if (kind == 0) val = xa[((long)(b * (LSEQ + 1) + 1 + i0)) * DINC + d];
                else {
                    float su = 0.f;
                    for (int j = i0; j < e0; ++j) su += xa[((long)(b * (LSEQ + 1) + 1 + j)) * DINC + d];
                    val = su / (float)(e0 - i0);
                }
            }
            ents[((long)(b * EMAXC + s)) * DINC + d] = val;
        }
        if (threadIdx.x == 0) {
            outPos[b * EMAXC + s]   = v ? (float)((kind == 0) ? (i0 + 1) : (e0 + 1)) : 0.f;
            outValid[b * EMAXC + s] = v ? 1.f : 0.f;
        }
    }
}

// =====================================================================
// seq = concat(ctx, ents) -> (528, 785)
// =====================================================================
__launch_bounds__(256)
__global__ void seqbuild_kernel(const float* __restrict__ xa, const float* __restrict__ ents,
                                float* __restrict__ seq)
{
    const int r = blockIdx.x;          // 0..527
    const int b = r / SEQT, s = r % SEQT;
    const float* src = (s == 0) ? (xa + (long)(b * (LSEQ + 1)) * DINC)
                                : (ents + (long)(b * EMAXC + (s - 1)) * DINC);
    for (int d = threadIdx.x; d < DINC; d += 256) seq[(long)r * DINC + d] = src[d];
}

// =====================================================================
// Attention: one block per (batch, head), S=33, dh=98
// =====================================================================
__launch_bounds__(128)
__global__ void attention_kernel(const float* __restrict__ qkv, float* __restrict__ out)
{
    const int b = blockIdx.x >> 3;
    const int h = blockIdx.x & 7;
    __shared__ float q[SEQT * DHH_], k[SEQT * DHH_], v[SEQT * DHH_], att[SEQT * SEQT];
    for (int t = threadIdx.x; t < SEQT * DHH_; t += 128) {
        int s = t / DHH_, d = t % DHH_;
        long rb = (long)(b * SEQT + s) * (3 * HDIMC) + h * DHH_ + d;
        q[t] = qkv[rb];
        k[t] = qkv[rb + HDIMC];
        v[t] = qkv[rb + 2 * HDIMC];
    }
    __syncthreads();
    const float scale = rsqrtf((float)DHH_);
    for (int t = threadIdx.x; t < SEQT * SEQT; t += 128) {
        int i = t / SEQT, j = t % SEQT;
        float s = 0.f;
        for (int d = 0; d < DHH_; ++d) s += q[i * DHH_ + d] * k[j * DHH_ + d];
        att[t] = s * scale;
    }
    __syncthreads();
    if (threadIdx.x < SEQT) {
        int i = threadIdx.x;
        float mx = -3.4e38f;
        for (int j = 0; j < SEQT; ++j) mx = fmaxf(mx, att[i * SEQT + j]);
        float su = 0.f;
        for (int j = 0; j < SEQT; ++j) { float e = __expf(att[i * SEQT + j] - mx); att[i * SEQT + j] = e; su += e; }
        float inv = 1.f / su;
        for (int j = 0; j < SEQT; ++j) att[i * SEQT + j] *= inv;
    }
    __syncthreads();
    for (int t = threadIdx.x; t < SEQT * DHH_; t += 128) {
        int i = t / DHH_, d = t % DHH_;
        float s = 0.f;
        for (int j = 0; j < SEQT; ++j) s += att[i * SEQT + j] * v[j * DHH_ + d];
        out[(long)(b * SEQT + i) * HDIMC + h * DHH_ + d] = s;
    }
}

// =====================================================================
// out = LayerNorm(h + t) * g + be     (one block per row, D = 784)
// =====================================================================
__launch_bounds__(256)
__global__ void add_ln_kernel(const float* __restrict__ h, const float* __restrict__ t,
                              const float* __restrict__ g, const float* __restrict__ be,
                              float* __restrict__ out, int D)
{
    __shared__ float y[HDIMC];
    __shared__ float red[256];
    const long r = blockIdx.x;
    float s = 0.f;
    for (int d = threadIdx.x; d < D; d += 256) { float vv = h[r * D + d] + t[r * D + d]; y[d] = vv; s += vv; }
    red[threadIdx.x] = s; __syncthreads();
    for (int o = 128; o > 0; o >>= 1) { if (threadIdx.x < o) red[threadIdx.x] += red[threadIdx.x + o]; __syncthreads(); }
    const float mean = red[0] / D; __syncthreads();
    s = 0.f;
    for (int d = threadIdx.x; d < D; d += 256) { float dv = y[d] - mean; s += dv * dv; }
    red[threadIdx.x] = s; __syncthreads();
    for (int o = 128; o > 0; o >>= 1) { if (threadIdx.x < o) red[threadIdx.x] += red[threadIdx.x + o]; __syncthreads(); }
    const float inv = rsqrtf(red[0] / D + 1e-5f);
    for (int d = threadIdx.x; d < D; d += 256) out[r * D + d] = (y[d] - mean) * inv * g[d] + be[d];
}

// =====================================================================
// enc copy to output + per-query |q|^2
// =====================================================================
__launch_bounds__(256)
__global__ void enccopy_kernel(const float* __restrict__ encAll, float* __restrict__ encOut,
                               float* __restrict__ qnorm)
{
    const int qr = blockIdx.x;               // 0..511
    const int b = qr >> 5, e = qr & 31;
    const float* src = encAll + (long)(b * SEQT + 1 + e) * NEDD;
    __shared__ float red[256];
    float s = 0.f;
    for (int d = threadIdx.x; d < NEDD; d += 256) { float v = src[d]; encOut[(long)qr * NEDD + d] = v; s += v * v; }
    red[threadIdx.x] = s; __syncthreads();
    for (int o = 128; o > 0; o >>= 1) { if (threadIdx.x < o) red[threadIdx.x] += red[threadIdx.x + o]; __syncthreads(); }
    if (threadIdx.x == 0) qnorm[qr] = red[0];
}

// =====================================================================
// KB prep: f32 -> bf16 (K padded to 320) + |kb|^2
// =====================================================================
__launch_bounds__(128)
__global__ void kbprep_kernel(const float* __restrict__ kb, __bf16* __restrict__ kbbf,
                              float* __restrict__ kbnorm)
{
    const long n = blockIdx.x;
    __shared__ float red[128];
    float s = 0.f;
    for (int d = threadIdx.x; d < KPAD; d += 128) {
        float v = (d < NEDD) ? kb[n * NEDD + d] : 0.f;
        kbbf[n * KPAD + d] = (__bf16)v;
        s += v * v;
    }
    red[threadIdx.x] = s; __syncthreads();
    for (int o = 64; o > 0; o >>= 1) { if (threadIdx.x < o) red[threadIdx.x] += red[threadIdx.x + o]; __syncthreads(); }
    if (threadIdx.x == 0) kbnorm[n] = red[0];
}

// =====================================================================
// TDM: DMA one 128-row x 320-col bf16 tile of kb into LDS.
// D# per CDNA5 ISA ch.8: group0 {count=1, lds_addr, global_addr, type=2},
// group1 {data_size=2B, tensor_dim0=320, tensor_dim1=100000,
//         tile 320x128, dim0_stride=320}.  2D tensor -> groups 2/3 zero.
// (6-arg builtin form: extra int32x8 group before cpol on this toolchain.)
// =====================================================================
__device__ __forceinline__ void tdm_load_kb_chunk(const __bf16* kbbf, unsigned ldsOff, int baseRow)
{
    unsigned long long ga = (unsigned long long)(size_t)kbbf
                          + (unsigned long long)baseRow * (KPAD * 2);
    v4u g0;
    g0[0] = 1u;                                             // count=1, user mode
    g0[1] = ldsOff;                                         // lds_addr (bytes)
    g0[2] = (unsigned)(ga & 0xFFFFFFFFu);                   // global_addr[31:0]
    g0[3] = (unsigned)((ga >> 32) & 0x01FFFFFFu) | (2u << 30); // global_addr[56:32] | type=2
    v8i_t g1;
    g1[0] = (int)(1u << 16);                                // data_size = 1 -> 2 bytes
    g1[1] = (int)(((unsigned)KPAD & 0xFFFFu) << 16);        // tensor_dim0[15:0] @ bits 63:48
    g1[2] = (int)(((unsigned)NKBC & 0xFFFFu) << 16);        // dim0 hi=0 | tensor_dim1[15:0]
    g1[3] = (int)(((unsigned)NKBC >> 16) | ((unsigned)KPAD << 16)); // dim1 hi | tile_dim0
    g1[4] = 128;                                            // tile_dim1=128 rows, tile_dim2=0
    g1[5] = KPAD;                                           // tensor_dim0_stride[31:0]
    g1[6] = 0;                                              // stride hi | dim1_stride lo
    g1[7] = 0;
    v4i  z4 = {0, 0, 0, 0};
    v8i_t z8 = {0, 0, 0, 0, 0, 0, 0, 0};
    __builtin_amdgcn_tensor_load_to_lds(g0, g1, z4, z4, z8, 0);
}

// =====================================================================
// Fused KNN: TDM double-buffers kb tiles into LDS; bf16 WMMA GEMM
// (16 queries x 128 KB rows per chunk) + per-thread top-10 + merge.
// block = 256 threads = 8 waves; grid = 32 blocks (512 queries / 16).
// =====================================================================
__launch_bounds__(256)
__global__ void knn_kernel(const float* __restrict__ encAll, const __bf16* __restrict__ kbbf,
                           const float* __restrict__ kbnorm, const float* __restrict__ qnorm,
                           int* __restrict__ idxOut)
{
    extern __shared__ char smem_knn[];
    float*  qs  = (float*)(smem_knn + KNN_QS_OFF);     // 16 x 320 f32
    __bf16* kbt = (__bf16*)(smem_knn + KNN_KBT_OFF);   // 2 x (128 x 320) bf16
    float*  d2s = (float*)(smem_knn + KNN_D2S_OFF);    // 16 x 128
    float*  md  = (float*)(smem_knn + KNN_MD_OFF);     // 16 x 160
    int*    mi_ = (int*)(smem_knn + KNN_MI_OFF);       // 16 x 160

    const int qbase = blockIdx.x * 16;
    for (int t = threadIdx.x; t < 16 * KPAD; t += 256) {
        int m = t / KPAD, d = t % KPAD;
        int qr = qbase + m; int b = qr >> 5, e = qr & 31;
        qs[t] = (d < NEDD) ? encAll[(long)(b * SEQT + 1 + e) * NEDD + d] : 0.f;
    }
    const int lane = threadIdx.x & 31;
    const int wave = threadIdx.x >> 5;
    const int g = lane >> 4;
    const int li = lane & 15;
    const int mq = threadIdx.x >> 4;        // query for update phase
    const int jj = threadIdx.x & 15;
    float td[KNNC]; int ti[KNNC];
#pragma unroll
    for (int k = 0; k < KNNC; ++k) { td[k] = 3.4e38f; ti[k] = 0x7fffffff; }

    const int NCHUNK = (NKBC + 127) / 128;            // 782
    if (wave == 0) tdm_load_kb_chunk(kbbf, KNN_KBT_OFF, 0);   // prime buffer 0
    __syncthreads();                                  // qs visible to all waves

    for (int c = 0; c < NCHUNK; ++c) {
        const int base = c * 128;
        if (wave == 0) {
            if (c + 1 < NCHUNK) {
                tdm_load_kb_chunk(kbbf, KNN_KBT_OFF + ((c + 1) & 1) * KNN_BUF_BYTES,
                                  (c + 1) * 128);
                __builtin_amdgcn_s_wait_tensorcnt(1);  // chunk c landed; c+1 in flight
            } else {
                __builtin_amdgcn_s_wait_tensorcnt(0);
            }
        }
        __syncthreads();                               // publish tile c
        const __bf16* kr = kbt + (size_t)(c & 1) * (128 * KPAD)
                               + (size_t)(wave * 16 + li) * KPAD;
        v8f acc = {};
        for (int k0 = 0; k0 < KPAD; k0 += 32) {
            v16bf a, bb;
#pragma unroll
            for (int j = 0; j < 8; ++j) {              // A frag (16-bit 16x32 layout)
                int kk = k0 + ((j < 4) ? (2 * j + 8 * g) : (16 + 2 * (j - 4) + 8 * g));
                a[2 * j]     = (__bf16)qs[li * KPAD + kk];
                a[2 * j + 1] = (__bf16)qs[li * KPAD + kk + 1];
            }
#pragma unroll
            for (int j = 0; j < 8; ++j) {              // B frag (32x16 layout) from LDS
                int kk = k0 + ((j < 4) ? (2 * j + 8 * g) : (16 + 2 * (j - 4) + 8 * g));
                bb[2 * j]     = kr[kk];
                bb[2 * j + 1] = kr[kk + 1];
            }
            acc = __builtin_amdgcn_wmma_f32_16x16x32_bf16(false, a, false, bb, (short)0, acc, false, false);
        }
#pragma unroll
        for (int i = 0; i < 8; ++i) {
            int m = i + 8 * g;
            int n = wave * 16 + li;
            int cand = base + n;
            d2s[m * 128 + n] = (cand < NKBC)
                ? (qnorm[qbase + m] - 2.f * acc[i] + kbnorm[cand]) : 3.4e38f;
        }
        __syncthreads();
#pragma unroll
        for (int s = 0; s < 8; ++s) {
            int n = jj + s * 16;
            int cand = base + n;
            float d = d2s[mq * 128 + n];
            if (cand < NKBC && d < td[KNNC - 1]) {
                int p = KNNC - 1;
                while (p > 0 && td[p - 1] > d) { td[p] = td[p - 1]; ti[p] = ti[p - 1]; --p; }
                td[p] = d; ti[p] = cand;
            }
        }
        __syncthreads();
    }
#pragma unroll
    for (int k = 0; k < KNNC; ++k) { md[(mq * 16 + jj) * KNNC + k] = td[k]; mi_[(mq * 16 + jj) * KNNC + k] = ti[k]; }
    __syncthreads();
    if (threadIdx.x < 16) {
        int m = threadIdx.x;
        for (int k = 0; k < KNNC; ++k) {
            float best = 3.4e38f; int bi = 0x7fffffff; int bp = -1;
            for (int p = 0; p < 160; ++p) {
                int off = (m * 16 + (p / KNNC)) * KNNC + (p % KNNC);
                float d = md[off]; int id = mi_[off];
                if (d < best || (d == best && id < bi)) { best = d; bi = id; bp = off; }
            }
            if (bp >= 0) md[bp] = 3.4e38f;
            idxOut[(qbase + m) * KNNC + k] = bi;
        }
    }
}

// =====================================================================
// Gather cand + build dist/ctx GEMM inputs + write cand to ned_out
// =====================================================================
__launch_bounds__(128)
__global__ void gatherprep_kernel(const float* __restrict__ kb, const int* __restrict__ idx,
                                  const float* __restrict__ encAll, float* __restrict__ nedOut,
                                  float* __restrict__ cand, float* __restrict__ din,
                                  float* __restrict__ cin)
{
    const int blk = blockIdx.x;          // 0..5119  (q*10+k)
    const int q = blk / KNNC;
    const int b = q >> 5, e = q & 31;
    const long kidx = idx[blk];
    const float* kr  = kb + kidx * NEDD;
    const float* enq = encAll + (long)(b * SEQT + 1 + e) * NEDD;
    const float* ctx = encAll + (long)(b * SEQT) * NEDD;
    for (int d = threadIdx.x; d < NEDD; d += 128) {
        float c = kr[d];
        cand[(long)blk * NEDD + d]         = c;
        din[(long)blk * NEDD + d]          = enq[d] - c;
        cin[(long)blk * NEDD + d]          = c * ctx[d];
        nedOut[(long)blk * (NEDD + 1) + 1 + d] = c;
    }
}

// =====================================================================
// row dot: out[r] = A[r,:] . w + b
// =====================================================================
__launch_bounds__(256)
__global__ void rowdot_kernel(const float* __restrict__ A, const float* __restrict__ w,
                              const float* __restrict__ bias, float* __restrict__ out, int K)
{
    __shared__ float red[256];
    const long r = blockIdx.x;
    float s = 0.f;
    for (int d = threadIdx.x; d < K; d += 256) s += A[r * K + d] * w[d];
    red[threadIdx.x] = s; __syncthreads();
    for (int o = 128; o > 0; o >>= 1) { if (threadIdx.x < o) red[threadIdx.x] += red[threadIdx.x + o]; __syncthreads(); }
    if (threadIdx.x == 0) out[r] = red[0] + bias[0];
}

// =====================================================================
// scores = ds + cs -> ned_out[...,0]; softmax over KNN; ent_full build
// =====================================================================
__launch_bounds__(128)
__global__ void combine_kernel(const float* __restrict__ ds, const float* __restrict__ cs,
                               const float* __restrict__ cand, const float* __restrict__ ents,
                               float* __restrict__ nedOut, float* __restrict__ entf)
{
    const int q = blockIdx.x;            // 0..511
    __shared__ float wv[KNNC];
    if (threadIdx.x == 0) {
        float mx = -3.4e38f;
        for (int k = 0; k < KNNC; ++k) {
            float s = ds[q * KNNC + k] + cs[q * KNNC + k];
            nedOut[(long)(q * KNNC + k) * (NEDD + 1)] = s;
            wv[k] = s; mx = fmaxf(mx, s);
        }
        float su = 0.f;
        for (int k = 0; k < KNNC; ++k) { wv[k] = __expf(wv[k] - mx); su += wv[k]; }
        float inv = 1.f / su;
        for (int k = 0; k < KNNC; ++k) wv[k] *= inv;
    }
    __syncthreads();
    for (int d = threadIdx.x; d < DINC; d += 128)
        entf[(long)q * D3C + d] = ents[(long)q * DINC + d];
    for (int d = threadIdx.x; d < NEDD; d += 128) {
        float s = 0.f;
        for (int k = 0; k < KNNC; ++k) s += wv[k] * cand[(long)(q * KNNC + k) * NEDD + d];
        entf[(long)q * D3C + DINC + d] = s;
    }
}

// =====================================================================
// Bilinear RE: re[b,e,f,r] = head_b @ bil_w[r] @ tail_b^T + bil_b[r] + lin
// block per (b,r); WMMA f32 both stages; weights staged in LDS (112 KB)
// =====================================================================
__launch_bounds__(128)
__global__ void bil_kernel(const float* __restrict__ head, const float* __restrict__ tail,
                           const float* __restrict__ bilw, const float* __restrict__ bilb,
                           const float* __restrict__ hlin, const float* __restrict__ tlin,
                           float* __restrict__ re)
{
    extern __shared__ float smem[];
    float* bw  = smem;                    // 128*128
    float* hd  = bw + HTC * HTC;          // 32*128
    float* tl  = hd + EMAXC * HTC;        // 32*128
    float* tmp = tl + EMAXC * HTC;        // 32*128
    const int b = blockIdx.x >> 5;
    const int r = blockIdx.x & 31;
    for (int t = threadIdx.x; t < HTC * HTC; t += 128) bw[t] = bilw[(long)r * HTC * HTC + t];
    for (int t = threadIdx.x; t < EMAXC * HTC; t += 128) {
        hd[t] = head[(long)(b * EMAXC) * HTC + t];
        tl[t] = tail[(long)(b * EMAXC) * HTC + t];
    }
    __syncthreads();
    const int lane = threadIdx.x & 31;
    const int wave = threadIdx.x >> 5;
    const int g = lane >> 4;
    const int li = lane & 15;
    // stage 1: tmp(32x128) = hd @ bw
    for (int tt = wave; tt < 16; tt += 4) {
        int tm = (tt >> 3) * 16;
        int tn = (tt & 7) * 16;
        v8f acc = {};
        for (int k0 = 0; k0 < HTC; k0 += 4) {
            int ka = k0 + 2 * g;
            v2f a, bb;
            a.x  = hd[(tm + li) * HTC + ka];     a.y  = hd[(tm + li) * HTC + ka + 1];
            bb.x = bw[ka * HTC + tn + li];       bb.y = bw[(ka + 1) * HTC + tn + li];
            acc = __builtin_amdgcn_wmma_f32_16x16x4_f32(false, a, false, bb, (short)0, acc, false, false);
        }
#pragma unroll
        for (int i = 0; i < 8; ++i) tmp[(tm + i + 8 * g) * HTC + tn + li] = acc[i];
    }
    __syncthreads();
    // stage 2: C(32x32) = tmp @ tl^T
    {
        int tm = (wave >> 1) * 16, tf = (wave & 1) * 16;
        v8f acc = {};
        for (int k0 = 0; k0 < HTC; k0 += 4) {
            int ka = k0 + 2 * g;
            v2f a, bb;
            a.x  = tmp[(tm + li) * HTC + ka];  a.y  = tmp[(tm + li) * HTC + ka + 1];
            bb.x = tl[(tf + li) * HTC + ka];   bb.y = tl[(tf + li) * HTC + ka + 1];
            acc = __builtin_amdgcn_wmma_f32_16x16x4_f32(false, a, false, bb, (short)0, acc, false, false);
        }
        const float b0 = bilb[r];
#pragma unroll
        for (int i = 0; i < 8; ++i) {
            int e = tm + i + 8 * g;
            int f = tf + li;
            float v = acc[i] + b0 + hlin[(b * EMAXC + e) * REDIM + r] + tlin[(b * EMAXC + f) * REDIM + r];
            re[(((long)(b * EMAXC + e)) * EMAXC + f) * REDIM + r] = v;
        }
    }
}

// =====================================================================
// Host side
// =====================================================================
static inline void gemm(const float* A, const float* B, const float* bias, float* C,
                        int M, int N, int K, int act, hipStream_t s)
{
    long tiles = (long)((M + 15) / 16) * ((N + 15) / 16);
    int blocks = (int)((tiles + 3) / 4);
    gemm_wmma_f32<<<blocks, 128, 0, s>>>(A, B, bias, C, M, N, K, act);
}

extern "C" void kernel_launch(void* const* d_in, const int* in_sizes, int n_in,
                              void* d_out, int out_size, void* d_ws, size_t ws_size,
                              hipStream_t stream)
{
    (void)in_sizes; (void)n_in; (void)out_size; (void)ws_size;
    const float* x       = (const float*)d_in[0];
    const float* ner_w0  = (const float*)d_in[1];
    const float* ner_b0  = (const float*)d_in[2];
    const float* ner_w   = (const float*)d_in[3];
    const float* ner_b   = (const float*)d_in[4];
    const float* ned1_w  = (const float*)d_in[5];
    const float* ned1_b  = (const float*)d_in[6];
    const float* tf_wqkv = (const float*)d_in[7];
    const float* tf_bqkv = (const float*)d_in[8];
    const float* tf_wo   = (const float*)d_in[9];
    const float* tf_bo   = (const float*)d_in[10];
    const float* tf_w1   = (const float*)d_in[11];
    const float* tf_b1   = (const float*)d_in[12];
    const float* tf_w2   = (const float*)d_in[13];
    const float* tf_b2   = (const float*)d_in[14];
    const float* tf_g1   = (const float*)d_in[15];
    const float* tf_be1  = (const float*)d_in[16];
    const float* tf_g2   = (const float*)d_in[17];
    const float* tf_be2  = (const float*)d_in[18];
    const float* ned_w   = (const float*)d_in[19];
    const float* ned_b   = (const float*)d_in[20];
    const float* kb      = (const float*)d_in[21];
    const float* dist1_w = (const float*)d_in[22];
    const float* dist1_b = (const float*)d_in[23];
    const float* dist2_w = (const float*)d_in[24];
    const float* dist2_b = (const float*)d_in[25];
    const float* ctx1_w  = (const float*)d_in[26];
    const float* ctx1_b  = (const float*)d_in[27];
    const float* ctx2_w  = (const float*)d_in[28];
    const float* ctx2_b  = (const float*)d_in[29];
    const float* h0_w    = (const float*)d_in[30];
    const float* h0_b    = (const float*)d_in[31];
    const float* h_w     = (const float*)d_in[32];
    const float* h_b     = (const float*)d_in[33];
    const float* t0_w    = (const float*)d_in[34];
    const float* t0_b    = (const float*)d_in[35];
    const float* t_w     = (const float*)d_in[36];
    const float* t_b     = (const float*)d_in[37];
    const float* bil_w   = (const float*)d_in[38];
    const float* bil_b   = (const float*)d_in[39];
    const float* relin_w = (const float*)d_in[40];

    float* out = (float*)d_out;
    const size_t off_ner  = 0;
    const size_t off_enc  = off_ner + (size_t)BZ * LSEQ * NERD;          // 139264
    const size_t off_ned  = off_enc + (size_t)BZ * EMAXC * NEDD;         // 292864
    const size_t off_pos  = off_ned + (size_t)BZ * EMAXC * KNNC * (NEDD + 1);
    const size_t off_val  = off_pos + (size_t)BZ * EMAXC;
    const size_t off_re   = off_val + (size_t)BZ * EMAXC;

    // ---- workspace bump allocator ----
    float* wsf = (float*)d_ws;
    size_t o = 0;
    auto A_ = [&](size_t n) { float* p = wsf + o; o += n; return p; };
    float* XA    = A_((size_t)NERROWS * DINC);
    float* H1    = A_((size_t)NERROWS * DB);
    float* NER_  = A_((size_t)NERROWS * NERD);
    float* ENTS  = A_((size_t)BZ * EMAXC * DINC);
    float* SEQ   = A_((size_t)ROWS * DINC);
    float* HB    = A_((size_t)ROWS * HDIMC);
    float* QKVB  = A_((size_t)ROWS * 3 * HDIMC);
    float* ATTO  = A_((size_t)ROWS * HDIMC);
    float* TMPB  = A_((size_t)ROWS * HDIMC);
    float* FFB   = A_((size_t)ROWS * FFC);
    float* ENCA  = A_((size_t)ROWS * NEDD);
    float* QN    = A_(512);
    float* KBN   = A_(NKBC);
    __bf16* KBBF = (__bf16*)A_((size_t)NKBC * KPAD / 2);   // bf16 storage
    int*   IDX   = (int*)A_(512 * KNNC);
    float* CAND  = A_((size_t)512 * KNNC * NEDD);
    float* DIN   = A_((size_t)512 * KNNC * NEDD);
    float* DH1   = A_((size_t)512 * KNNC * NEDD);
    float* DSV   = A_(512 * KNNC);
    float* CIN   = A_((size_t)512 * KNNC * NEDD);
    float* CH1   = A_((size_t)512 * KNNC * NEDD);
    float* CSV   = A_(512 * KNNC);
    float* ENTF  = A_((size_t)512 * D3C);
    float* HHB   = A_((size_t)512 * D3C);
    float* HEAD  = A_((size_t)512 * HTC);
    float* TLH   = A_((size_t)512 * D3C);
    float* TAIL  = A_((size_t)512 * HTC);
    float* HLIN  = A_((size_t)512 * REDIM);
    float* TLIN  = A_((size_t)512 * REDIM);

    // ---- 1. NER head ----
    gemm(x, ner_w0, ner_b0, H1, NERROWS, DB, DB, 1, stream);
    gemm(H1, ner_w, ner_b, NER_, NERROWS, NERD, DB, 0, stream);
    xabuild_kernel<<<NERROWS, 256, 0, stream>>>(x, NER_, XA, out + off_ner);

    // ---- 2. entity filter + seq ----
    entity_filter_kernel<<<BZ, 512, 0, stream>>>(XA, ENTS, out + off_pos, out + off_val);
    seqbuild_kernel<<<ROWS, 256, 0, stream>>>(XA, ENTS, SEQ);
    gemm(SEQ, ned1_w, ned1_b, HB, ROWS, HDIMC, DINC, 1, stream);

    // ---- 3. transformer (6 layers) ----
    for (int l = 0; l < NLAY; ++l) {
        gemm(HB, tf_wqkv + (size_t)l * HDIMC * 3 * HDIMC, tf_bqkv + (size_t)l * 3 * HDIMC,
             QKVB, ROWS, 3 * HDIMC, HDIMC, 0, stream);
        attention_kernel<<<BZ * NHEADC, 128, 0, stream>>>(QKVB, ATTO);
        gemm(ATTO, tf_wo + (size_t)l * HDIMC * HDIMC, tf_bo + (size_t)l * HDIMC,
             TMPB, ROWS, HDIMC, HDIMC, 0, stream);
        add_ln_kernel<<<ROWS, 256, 0, stream>>>(HB, TMPB, tf_g1 + (size_t)l * HDIMC,
                                                tf_be1 + (size_t)l * HDIMC, HB, HDIMC);
        gemm(HB, tf_w1 + (size_t)l * HDIMC * FFC, tf_b1 + (size_t)l * FFC,
             FFB, ROWS, FFC, HDIMC, 1, stream);
        gemm(FFB, tf_w2 + (size_t)l * FFC * HDIMC, tf_b2 + (size_t)l * HDIMC,
             TMPB, ROWS, HDIMC, FFC, 0, stream);
        add_ln_kernel<<<ROWS, 256, 0, stream>>>(HB, TMPB, tf_g2 + (size_t)l * HDIMC,
                                                tf_be2 + (size_t)l * HDIMC, HB, HDIMC);
    }

    // ---- 4. encoding + KNN over 100k KB (TDM + bf16 WMMA, fused top-k) ----
    gemm(HB, ned_w, ned_b, ENCA, ROWS, NEDD, HDIMC, 0, stream);
    enccopy_kernel<<<512, 256, 0, stream>>>(ENCA, out + off_enc, QN);
    kbprep_kernel<<<NKBC, 128, 0, stream>>>(kb, KBBF, KBN);
    knn_kernel<<<32, 256, KNN_SHMEM, stream>>>(ENCA, KBBF, KBN, QN, IDX);

    // ---- 5. candidate scoring ----
    gatherprep_kernel<<<512 * KNNC, 128, 0, stream>>>(kb, IDX, ENCA, out + off_ned, CAND, DIN, CIN);
    gemm(DIN, dist1_w, dist1_b, DH1, 512 * KNNC, NEDD, NEDD, 1, stream);
    rowdot_kernel<<<512 * KNNC, 256, 0, stream>>>(DH1, dist2_w, dist2_b, DSV, NEDD);
    gemm(CIN, ctx1_w, ctx1_b, CH1, 512 * KNNC, NEDD, NEDD, 1, stream);
    rowdot_kernel<<<512 * KNNC, 256, 0, stream>>>(CH1, ctx2_w, ctx2_b, CSV, NEDD);
    combine_kernel<<<512, 128, 0, stream>>>(DSV, CSV, CAND, ENTS, out + off_ned, ENTF);

    // ---- 6. relation extraction ----
    gemm(ENTF, h0_w, h0_b, HHB, 512, D3C, D3C, 1, stream);
    gemm(HHB, h_w, h_b, HEAD, 512, HTC, D3C, 0, stream);
    gemm(ENTF, t0_w, t0_b, TLH, 512, D3C, D3C, 1, stream);
    gemm(TLH, t_w, t_b, TAIL, 512, HTC, D3C, 0, stream);
    gemm(HEAD, relin_w,               nullptr, HLIN, 512, REDIM, HTC, 0, stream);
    gemm(TAIL, relin_w + HTC * REDIM, nullptr, TLIN, 512, REDIM, HTC, 0, stream);
    const size_t bilShmem = (size_t)(HTC * HTC + 3 * EMAXC * HTC) * sizeof(float); // 112 KB
    bil_kernel<<<BZ * REDIM, 128, bilShmem, stream>>>(HEAD, TAIL, bil_w, bil_b, HLIN, TLIN, out + off_re);
}
